// BasicBlockBi_30305289240892
// MI455X (gfx1250) — 8-bit matrix operand order repaired and checked by reading, compiled, not run on hardware
//
#include <hip/hip_runtime.h>

typedef __attribute__((ext_vector_type(8))) int v8i;

#define BN_EPS 1e-5f
#define TERN_FRAC 0.7f

__global__ __launch_bounds__(64)
void zerofill_kernel(int4* __restrict__ p) {
    if (threadIdx.x < 16) {
        int4 z; z.x = 0; z.y = 0; z.z = 0; z.w = 0;
        p[threadIdx.x] = z;
    }
}

__global__ __launch_bounds__(256)
void bnprep_kernel(const float* __restrict__ gamma, const float* __restrict__ beta,
                   const float* __restrict__ mean,  const float* __restrict__ var,
                   float* __restrict__ scale, float* __restrict__ bias) {
    int c = threadIdx.x;
    float inv = gamma[c] * rsqrtf(var[c] + BN_EPS);
    scale[c] = inv;
    bias[c]  = beta[c] - mean[c] * inv;
}

__global__ __launch_bounds__(256)
void absmean_kernel(const float* __restrict__ x, const float* __restrict__ scale,
                    const float* __restrict__ bias, float* __restrict__ partials, int n) {
    __shared__ float sdata[256];
    float s = 0.f;
    for (int i = blockIdx.x * blockDim.x + threadIdx.x; i < n; i += gridDim.x * blockDim.x) {
        int c = (i >> 10) & 255;
        float v = x[i] * scale[c] + bias[c];
        s += fabsf(v);
    }
    sdata[threadIdx.x] = s;
    __syncthreads();
    for (int off = 128; off > 0; off >>= 1) {
        if (threadIdx.x < off) sdata[threadIdx.x] += sdata[threadIdx.x + off];
        __syncthreads();
    }
    if (threadIdx.x == 0) partials[blockIdx.x] = sdata[0];
}

__global__ __launch_bounds__(256)
void delta_kernel(const float* __restrict__ partials, float* __restrict__ delta,
                  int nblocks, float inv_numel) {
    __shared__ float sdata[256];
    float s = 0.f;
    for (int i = threadIdx.x; i < nblocks; i += 256) s += partials[i];
    sdata[threadIdx.x] = s;
    __syncthreads();
    for (int off = 128; off > 0; off >>= 1) {
        if (threadIdx.x < off) sdata[threadIdx.x] += sdata[threadIdx.x + off];
        __syncthreads();
    }
    if (threadIdx.x == 0) *delta = TERN_FRAC * sdata[0] * inv_numel;
}

__global__ __launch_bounds__(256)
void quant_kernel(const float* __restrict__ x, const float* __restrict__ scale,
                  const float* __restrict__ bias, const float* __restrict__ delta_p,
                  signed char* __restrict__ xq) {
    __shared__ signed char tile[8192];
    int row = blockIdx.x;
    int b = row >> 5, y = row & 31;
    float delta = *delta_p;
    const float* xb = x + ((size_t)b << 18) + y * 32;
    for (int e = threadIdx.x; e < 8192; e += 256) {
        int c = e >> 5, xx = e & 31;
        float v = xb[c * 1024 + xx] * scale[c] + bias[c];
        signed char q = (v > delta) ? (signed char)1 : ((v < -delta) ? (signed char)-1 : (signed char)0);
        tile[xx * 256 + c] = q;
    }
    __syncthreads();
    int4* dst = (int4*)(xq + ((size_t)row << 13));
    const int4* src = (const int4*)tile;
    for (int e = threadIdx.x; e < 512; e += 256) dst[e] = src[e];
}

__global__ __launch_bounds__(256)
void wprep_kernel(const float* __restrict__ w, signed char* __restrict__ wsign,
                  float* __restrict__ alpha) {
    __shared__ float sdata[256];
    int o = blockIdx.x;
    int i = threadIdx.x;
    const float* wp = w + ((size_t)o * 256 + i) * 9;
    float s = 0.f;
    #pragma unroll
    for (int t = 0; t < 9; ++t) {
        float v = wp[t];
        s += fabsf(v);
        signed char q = (v > 0.f) ? (signed char)1 : ((v < 0.f) ? (signed char)-1 : (signed char)0);
        wsign[(((size_t)t * 256 + o) << 8) + i] = q;
    }
    sdata[i] = s;
    __syncthreads();
    for (int off = 128; off > 0; off >>= 1) {
        if (i < off) sdata[i] += sdata[i + off];
        __syncthreads();
    }
    if (i == 0) alpha[o] = sdata[0] * (1.0f / 2304.0f);
}

__global__ __launch_bounds__(256)
void tbnconv_kernel(const signed char* __restrict__ xq,
                    const signed char* __restrict__ wsign,
                    const float* __restrict__ alpha,
                    const float* resid, float* out,
                    unsigned zoff) {
    int wave  = (blockIdx.x << 3) + (threadIdx.x >> 5);
    int lane  = threadIdx.x & 31;
    int ntile = wave & 15;
    int mtile = wave >> 4;
    int b  = mtile >> 6;
    int y  = (mtile >> 1) & 31;
    int x0 = (mtile & 1) << 4;
    int half = lane >> 4;
    int l15  = lane & 15;
    int n    = (ntile << 4) + l15;
    int xm   = x0 + l15;

    unsigned aoff[9];
    #pragma unroll
    for (int t = 0; t < 9; ++t) {
        const int dy = t / 3 - 1, dx = t % 3 - 1;
        int yy = y + dy;
        int xx = xm + dx;
        bool v = (yy >= 0) && (yy < 32) && (xx >= 0) && (xx < 32);
        unsigned real = ((unsigned)((b * 32 + yy) * 32 + xx)) << 8;
        aoff[t] = (v ? real : zoff) + (unsigned)(half << 4);
    }
    const unsigned boff = ((unsigned)n << 8) + (unsigned)(half << 4);

    #pragma unroll
    for (int t = 0; t < 9; ++t)
        __builtin_prefetch(wsign + (size_t)boff + (unsigned)(t << 16), 0, 1);

    auto load_frags = [&](int s, v8i& a, v8i& bf) {
        const int t  = s >> 2;
        const int kc = (s & 3) << 6;
        const signed char* pa = xq + (size_t)aoff[t];
        int2 q0 = *(const int2*)(pa + kc);
        int2 q1 = *(const int2*)(pa + kc + 8);
        int2 q2 = *(const int2*)(pa + kc + 32);
        int2 q3 = *(const int2*)(pa + kc + 40);
        a[0] = q0.x; a[1] = q0.y; a[2] = q1.x; a[3] = q1.y;
        a[4] = q2.x; a[5] = q2.y; a[6] = q3.x; a[7] = q3.y;
        const signed char* pb = wsign + (size_t)boff;
        const int woff = (t << 16) + kc;
        int4 b0 = *(const int4*)(pb + woff);
        int4 b1 = *(const int4*)(pb + woff + 32);
        bf[0] = b0.x; bf[1] = b0.y; bf[2] = b0.z; bf[3] = b0.w;
        bf[4] = b1.x; bf[5] = b1.y; bf[6] = b1.z; bf[7] = b1.w;
    };

    v8i acc0 = {}, acc1 = {};
    v8i ca, cb;
    load_frags(0, ca, cb);

    #pragma unroll
    for (int s = 0; s < 36; ++s) {
        v8i na = ca, nb = cb;
        if (s + 1 < 36) load_frags(s + 1, na, nb);
        if (s & 1)
            acc1 = __builtin_amdgcn_wmma_i32_16x16x64_iu8(true, ca, true, cb, acc1,
                                                          false, false);
        else
            acc0 = __builtin_amdgcn_wmma_i32_16x16x64_iu8(true, ca, true, cb, acc0,
                                                          false, false);
        ca = na; cb = nb;
    }

    float al = alpha[n];
    size_t base = (((size_t)(b * 256 + n) * 32) + y) * 32 + x0 + (half << 3);
    float4 r0 = *(const float4*)(resid + base);
    float4 r1 = *(const float4*)(resid + base + 4);
    float4 o0, o1;
    o0.x = r0.x + al * (float)(acc0[0] + acc1[0]);
    o0.y = r0.y + al * (float)(acc0[1] + acc1[1]);
    o0.z = r0.z + al * (float)(acc0[2] + acc1[2]);
    o0.w = r0.w + al * (float)(acc0[3] + acc1[3]);
    o1.x = r1.x + al * (float)(acc0[4] + acc1[4]);
    o1.y = r1.y + al * (float)(acc0[5] + acc1[5]);
    o1.z = r1.z + al * (float)(acc0[6] + acc1[6]);
    o1.w = r1.w + al * (float)(acc0[7] + acc1[7]);
    *(float4*)(out + base)     = o0;
    *(float4*)(out + base + 4) = o1;
}

extern "C" void kernel_launch(void* const* d_in, const int* in_sizes, int n_in,
                              void* d_out, int out_size, void* d_ws, size_t ws_size,
                              hipStream_t stream) {
    const float* x      = (const float*)d_in[0];
    const float* w1     = (const float*)d_in[1];
    const float* w2     = (const float*)d_in[2];
    const float* gamma1 = (const float*)d_in[3];
    const float* beta1  = (const float*)d_in[4];
    const float* mean1  = (const float*)d_in[5];
    const float* var1   = (const float*)d_in[6];
    const float* gamma2 = (const float*)d_in[7];
    const float* beta2  = (const float*)d_in[8];
    const float* mean2  = (const float*)d_in[9];
    const float* var2   = (const float*)d_in[10];

    const int NELEM = 32 * 256 * 32 * 32;

    char* ws = (char*)d_ws;
    signed char* xq     = (signed char*)ws;
    signed char* wsign1 = (signed char*)(ws + 8388608);
    signed char* wsign2 = (signed char*)(ws + 8978432);
    const unsigned ZOFF = 9568256;
    signed char* zpage  = (signed char*)(ws + ZOFF);
    float* scale1   = (float*)(ws + 9568512);
    float* bias1    = scale1 + 256;
    float* scale2   = scale1 + 512;
    float* bias2    = scale1 + 768;
    float* alpha1   = scale1 + 1024;
    float* alpha2   = scale1 + 1280;
    float* partials = scale1 + 1536;
    float* delta1   = partials + 1024;
    float* delta2   = delta1 + 1;

    float* h   = (float*)d_out;
    float* out = (float*)d_out;

    zerofill_kernel<<<1, 64, 0, stream>>>((int4*)zpage);
    bnprep_kernel<<<1, 256, 0, stream>>>(gamma1, beta1, mean1, var1, scale1, bias1);
    bnprep_kernel<<<1, 256, 0, stream>>>(gamma2, beta2, mean2, var2, scale2, bias2);
    wprep_kernel<<<256, 256, 0, stream>>>(w1, wsign1, alpha1);
    wprep_kernel<<<256, 256, 0, stream>>>(w2, wsign2, alpha2);

    absmean_kernel<<<1024, 256, 0, stream>>>(x, scale1, bias1, partials, NELEM);
    delta_kernel<<<1, 256, 0, stream>>>(partials, delta1, 1024, 1.0f / (float)NELEM);
    quant_kernel<<<1024, 256, 0, stream>>>(x, scale1, bias1, delta1, xq);
    tbnconv_kernel<<<4096, 256, 0, stream>>>(xq, wsign1, alpha1, x, h, ZOFF);

    absmean_kernel<<<1024, 256, 0, stream>>>(h, scale2, bias2, partials, NELEM);
    delta_kernel<<<1, 256, 0, stream>>>(partials, delta2, 1024, 1.0f / (float)NELEM);
    quant_kernel<<<1024, 256, 0, stream>>>(h, scale2, bias2, delta2, xq);
    tbnconv_kernel<<<4096, 256, 0, stream>>>(xq, wsign2, alpha2, h, out, ZOFF);
}
